// DTransformerLayer_38654705664447
// MI455X (gfx1250) — matrix operand order repaired and checked by reading, not compiled, not run on hardware
//
#include <hip/hip_runtime.h>

// ---------------------------------------------------------------------------
// DTransformer layer forward for MI455X (gfx1250, wave32, WMMA).
//   B=8, S=1024, D=512, H=8, DK=64.
//   d_out = y [B,S,D] fp32  ++  scores [B,H,S,S] fp32.
//
// Data-movement plan (23.3 TB/s HBM, 192 MB L2, 320 KB LDS/WGP):
//  * all GEMMs on v_wmma_f32_16x16x32_f16 (f16 A/B, f32 accumulate)
//  * weights pre-transposed to k-major f16 so B fragments = 2 x b128 loads
//  * V stored [B,H,DK,S] so PV B fragments = 2 x b128 loads
//  * projection weight tile double-buffered in LDS via
//    global_load_async_to_lds_b128; s_wait_asynccnt 0x1 overlaps the next
//    tile's copy with the current tile's WMMA chain (async ops complete
//    in order, so cnt<=1 implies the current tile has landed)
// ---------------------------------------------------------------------------

#define B_  8
#define S_  1024
#define D_  512
#define H_  8
#define DK_ 64
#define NEGC (-1e32f)
#define WPAD 40   // padded LDS row stride in halves (80 B, multiple of 16 B)

typedef __attribute__((ext_vector_type(16))) _Float16 v16h;
typedef __attribute__((ext_vector_type(8)))  _Float16 v8h;
typedef __attribute__((ext_vector_type(8)))  float    v8f;

// --- wave32 reductions ------------------------------------------------------
__device__ __forceinline__ float wave_max(float v) {
  #pragma unroll
  for (int o = 16; o > 0; o >>= 1) v = fmaxf(v, __shfl_xor(v, o, 32));
  return v;
}
__device__ __forceinline__ float wave_sum(float v) {
  #pragma unroll
  for (int o = 16; o > 0; o >>= 1) v += __shfl_xor(v, o, 32);
  return v;
}

// --- fragment helpers -------------------------------------------------------
// A 16x32 f16 fragment (ISA 7.12.2): lane half h holds two contiguous 8-half
// runs at k = h*8 and k = 16 + h*8.  B 32x16: same two 8-half runs at
// k = h*8 and k = 16 + h*8.  With k-major storage each run is a single 16-byte load.
__device__ __forceinline__ v16h pack16(v8h lo, v8h hi) {
  v16h r;
  #pragma unroll
  for (int e = 0; e < 8; ++e) { r[e] = lo[e]; r[e + 8] = hi[e]; }
  return r;
}
__device__ __forceinline__ v16h load_a_f32(const float* base, int half) {
  // base points at row start + kk; convert two 8-float runs to f16
  v16h a;
  #pragma unroll
  for (int r2 = 0; r2 < 2; ++r2) {
    const float4* p = (const float4*)(base + r2 * 16 + half * 8);
    float4 f0 = p[0], f1 = p[1];
    int o = r2 * 8;
    a[o + 0] = (_Float16)f0.x; a[o + 1] = (_Float16)f0.y;
    a[o + 2] = (_Float16)f0.z; a[o + 3] = (_Float16)f0.w;
    a[o + 4] = (_Float16)f1.x; a[o + 5] = (_Float16)f1.y;
    a[o + 6] = (_Float16)f1.z; a[o + 7] = (_Float16)f1.w;
  }
  return a;
}
__device__ __forceinline__ v16h load_a_f16(const _Float16* base, int half) {
  v8h lo = *(const v8h*)(base + half * 8);
  v8h hi = *(const v8h*)(base + 16 + half * 8);
  return pack16(lo, hi);
}
__device__ __forceinline__ v16h load_b_kmajor(const _Float16* base, int half) {
  v8h lo = *(const v8h*)(base + half * 8);
  v8h hi = *(const v8h*)(base + 16 + half * 8);
  return pack16(lo, hi);
}

// --- CDNA5 async global->LDS copy (CDNA5 ISA) --------------
typedef __attribute__((address_space(3))) _Float16 lds_h;
__device__ __forceinline__ unsigned lds_off(const _Float16* p) {
  return (unsigned)(uintptr_t)(lds_h*)p;
}
__device__ __forceinline__ void async_copy_b128(unsigned loff, const void* g) {
  asm volatile("global_load_async_to_lds_b128 %0, %1, off"
               :: "v"(loff), "v"(g) : "memory");
}
__device__ __forceinline__ void wait_async_le1() {   // current tile landed
  asm volatile("s_wait_asynccnt 0x1" ::: "memory");
}
__device__ __forceinline__ void wait_async_le0() {   // everything landed
  asm volatile("s_wait_asynccnt 0x0" ::: "memory");
}

// ---------------------------------------------------------------------------
// Kernel 0: fp32 -> f16 transpose-convert for weights: WT[n][k] = W[k][n]
// ---------------------------------------------------------------------------
__global__ void cvtT_kernel(const float* __restrict__ src,
                            _Float16* __restrict__ dst) {
  int idx = blockIdx.x * blockDim.x + threadIdx.x;   // coalesced read
  int k = idx >> 9, n = idx & (D_ - 1);
  dst[n * D_ + k] = (_Float16)src[idx];
}

// ---------------------------------------------------------------------------
// Kernel 1: projection GEMM  Y = f16( X[M,512] @ W + b ), W given k-major.
//   Block 256 thr (8 waves) -> 128 rows x 64 cols; wave = 16x64 strip.
//   Weight tile (64n x 32k, shared by all waves) double-buffered via
//   async->LDS.  TRANSPOSE=0: Y[B,H,S,DK]   TRANSPOSE=1: Y[B,H,DK,S]
// ---------------------------------------------------------------------------
template <bool TRANSPOSE>
__global__ void proj_kernel(const float* __restrict__ X,
                            const _Float16* __restrict__ WT,  // [n][k]
                            const float* __restrict__ bias,
                            _Float16* __restrict__ Y) {
  __shared__ _Float16 Ws[2][64 * WPAD];              // 2 x 5 KB weight tiles
  const int lane = threadIdx.x & 31, wave = threadIdx.x >> 5;
  const int half = lane >> 4, ln = lane & 15;
  const int r0 = blockIdx.y * 128 + wave * 16;
  const int n0 = blockIdx.x * 64;
  const int row = r0 + ln;

  // async staging assignment: thread t copies 8 halves (16 B)
  const int sn = threadIdx.x >> 2;                   // 0..63 (tile n-row)
  const int sk = (threadIdx.x & 3) * 8;              // 0,8,16,24

  v8f acc[4] = {v8f{}, v8f{}, v8f{}, v8f{}};

  // prologue: stage first weight tile
  async_copy_b128(lds_off(&Ws[0][sn * WPAD + sk]),
                  WT + (size_t)(n0 + sn) * D_ + sk);

  for (int kk = 0; kk < D_; kk += 32) {
    const int cur = (kk >> 5) & 1;
    const bool more = (kk + 32) < D_;
    if (more)                                        // overlap next copy
      async_copy_b128(lds_off(&Ws[cur ^ 1][sn * WPAD + sk]),
                      WT + (size_t)(n0 + sn) * D_ + kk + 32 + sk);
    // A fragment straight from global fp32 (4 x float4 + cvt)
    v16h a = load_a_f32(X + row * D_ + kk, half);
    if (more) wait_async_le1(); else wait_async_le0();
    __syncthreads();                                 // Ws[cur] visible to all
    #pragma unroll
    for (int t = 0; t < 4; ++t) {
      v16h b = load_b_kmajor(&Ws[cur][(t * 16 + ln) * WPAD], half);
      acc[t] = __builtin_amdgcn_wmma_f32_16x16x32_f16(
          false, a, false, b, (short)0, acc[t], false, false);
    }
    __syncthreads();                                 // reads done -> reusable
  }

  #pragma unroll
  for (int t = 0; t < 4; ++t) {
    #pragma unroll
    for (int v = 0; v < 8; ++v) {
      int m = v + half * 8;
      int grow = r0 + m;                             // [0, B*S)
      int col  = n0 + t * 16 + ln;                   // [0, D)
      float val = acc[t][v] + bias[col];
      int b = grow >> 10, s = grow & (S_ - 1);
      int h = col >> 6,  dk = col & (DK_ - 1);
      if (TRANSPOSE)
        Y[(((b * H_ + h) * DK_ + dk) << 10) + s] = (_Float16)val;
      else
        Y[(((b * H_ + h) << 10) + s) * DK_ + dk] = (_Float16)val;
    }
  }
}

// ---------------------------------------------------------------------------
// Kernel 2: attention for one (b,h, 16-row tile).  Block 128 thr (4 waves).
//   Dynamic LDS: 16 x 1024 fp32 score rows (64 KB).
// ---------------------------------------------------------------------------
__global__ void attn_kernel(const _Float16* __restrict__ q16,  // [B,H,S,DK]
                            const _Float16* __restrict__ k16,  // [B,H,S,DK]
                            const _Float16* __restrict__ vT16, // [B,H,DK,S]
                            const float* __restrict__ gammas,
                            float* __restrict__ scores_out,
                            _Float16* __restrict__ attn16) {
  extern __shared__ float sc[];                      // [16][1024]
  const int lane = threadIdx.x & 31, wave = threadIdx.x >> 5;  // 4 waves
  const int half = lane >> 4, ln = lane & 15;
  const int it = blockIdx.x & 63;
  const int bh = blockIdx.x >> 6;                    // b*H + h
  const int h  = bh & (H_ - 1);
  const int i0 = it * 16;
  const _Float16* qb  = q16  + (size_t)bh * S_ * DK_;
  const _Float16* kb  = k16  + (size_t)bh * S_ * DK_;
  const _Float16* vbT = vT16 + (size_t)bh * S_ * DK_;

  // Q fragments: rows i0..i0+15, K-dim 64 = 2 x 32 (two b128 loads each)
  v16h aq[2];
  #pragma unroll
  for (int kq = 0; kq < 2; ++kq)
    aq[kq] = load_a_f16(qb + (i0 + ln) * DK_ + kq * 32, half);

  const float rscale = 0.125f;                       // 1/sqrt(DK)

  // --- phase 1: QK^T tiles (waves interleave column tiles) ---
  for (int jt = wave; jt < 64; jt += 4) {
    if (jt <= it) {
      v8f c = {};
      #pragma unroll
      for (int kq = 0; kq < 2; ++kq) {
        // B[d,j] = k[j,d]: per lane j = jt*16+ln fixed, d contiguous
        v16h bf = load_b_kmajor(kb + (jt * 16 + ln) * DK_ + kq * 32, half);
        c = __builtin_amdgcn_wmma_f32_16x16x32_f16(
            false, aq[kq], false, bf, (short)0, c, false, false);
      }
      #pragma unroll
      for (int v = 0; v < 8; ++v) {
        int m = v + half * 8;
        int j = jt * 16 + ln;
        sc[m * S_ + j] = (j < i0 + m) ? c[v] * rscale : NEGC;  // strict causal
      }
    } else {
      #pragma unroll
      for (int v = 0; v < 8; ++v)
        sc[(v + half * 8) * S_ + jt * 16 + ln] = NEGC;
    }
  }
  __syncthreads();

  const float g = -fabsf(gammas[h]);

  // --- phase 2: per-row softmax / decay / softmax / maxout ---
  for (int r = wave; r < 16; r += 4) {
    float* row = sc + r * S_;
    const int irow = i0 + r;
    const float fi = (float)irow;

    float m1 = -3.4e38f;
    for (int t = 0; t < 32; ++t) m1 = fmaxf(m1, row[t * 32 + lane]);
    m1 = wave_max(m1);
    float csum = 0.f;                                // contiguous chunk/lane
    for (int t = 0; t < 32; ++t) csum += __expf(row[lane * 32 + t] - m1);
    float total = wave_sum(csum);
    float pre = csum;                                // exclusive lane prefix
    #pragma unroll
    for (int o = 1; o < 32; o <<= 1) {
      float u = __shfl_up(pre, o, 32);
      if (lane >= o) pre += u;
    }
    pre -= csum;
    float run = pre, inv_total = 1.0f / total;       // disttotal == 1
    for (int t = 0; t < 32; ++t) {
      int j = lane * 32 + t;
      run += __expf(row[j] - m1);
      float rem  = 1.0f - run * inv_total;
      float pos  = fabsf(fi - (float)j);
      float dist = sqrtf(fmaxf(rem * pos, 0.0f));
      float eff  = fminf(fmaxf(__expf(g * dist), 1e-5f), 1e5f);
      row[j] = (j >= irow) ? NEGC : row[j] * eff;
    }
    float m2 = -3.4e38f;
    for (int t = 0; t < 32; ++t) m2 = fmaxf(m2, row[t * 32 + lane]);
    m2 = wave_max(m2);
    float s2 = 0.f;
    for (int t = 0; t < 32; ++t) s2 += __expf(row[t * 32 + lane] - m2);
    s2 = wave_sum(s2);
    float inv_s2 = 1.0f / s2;
    float fm = 0.f;
    for (int t = 0; t < 32; ++t) {
      int j = t * 32 + lane;
      float p = (j >= irow) ? 0.0f : __expf(row[j] - m2) * inv_s2;
      row[j] = p;
      fm = fmaxf(fm, p);
    }
    fm = wave_max(fm);
    float scale = fminf(1.0f / (fm + 1e-8f), 5.0f);
    float* srow = scores_out + ((size_t)bh * S_ + irow) * S_;
    for (int t = 0; t < 32; ++t) {
      int j = t * 32 + lane;
      float p = row[j] * scale;
      row[j] = p;
      srow[j] = p;
    }
  }
  __syncthreads();

  // --- phase 3: attn = scores[16x1024] @ V; wave w owns one 16-col tile ---
  v8f acc = {};
  const int dk = wave * 16 + ln;                     // B-fragment column
  for (int kk = 0; kk < S_; kk += 32) {
    v16h a = load_a_f32(sc + ln * S_ + kk, half);    // fp32 LDS -> f16 frag
    v16h bf = load_b_kmajor(vbT + dk * S_ + kk, half); // V^T: contiguous in s
    acc = __builtin_amdgcn_wmma_f32_16x16x32_f16(
        false, a, false, bf, (short)0, acc, false, false);
  }
  const int b = bh >> 3;
  #pragma unroll
  for (int v = 0; v < 8; ++v) {
    int m = v + half * 8;
    int col = h * DK_ + wave * 16 + ln;              // concat layout [B,S,D]
    attn16[((size_t)(b * S_ + i0 + m)) * D_ + col] = (_Float16)acc[v];
  }
}

// ---------------------------------------------------------------------------
// Kernel 3: out = attn16 @ Wo + bo; x = query + out; LayerNorm -> y.
//   Block 256 thr (8 waves) owns 16 full rows; Wo given k-major.
// ---------------------------------------------------------------------------
__global__ void outproj_ln_kernel(const _Float16* __restrict__ A,   // [M,512]
                                  const _Float16* __restrict__ WoT, // [n][k]
                                  const float* __restrict__ bo,
                                  const float* __restrict__ query,
                                  const float* __restrict__ ln_w,
                                  const float* __restrict__ ln_b,
                                  float* __restrict__ Yout) {
  __shared__ float xs[16][D_];                       // 32 KB
  const int lane = threadIdx.x & 31, wave = threadIdx.x >> 5;
  const int half = lane >> 4, ln = lane & 15;
  const int r0 = blockIdx.x * 16;
  const int n0 = wave * 64;
  const int row = r0 + ln;

  v8f acc[4] = {v8f{}, v8f{}, v8f{}, v8f{}};
  for (int kk = 0; kk < D_; kk += 32) {
    v16h a = load_a_f16(A + row * D_ + kk, half);    // two b128 loads
    #pragma unroll
    for (int t = 0; t < 4; ++t) {
      v16h b = load_b_kmajor(WoT + (size_t)(n0 + t * 16 + ln) * D_ + kk, half);
      acc[t] = __builtin_amdgcn_wmma_f32_16x16x32_f16(
          false, a, false, b, (short)0, acc[t], false, false);
    }
  }
  #pragma unroll
  for (int t = 0; t < 4; ++t) {
    #pragma unroll
    for (int v = 0; v < 8; ++v) {
      int m = v + half * 8;
      int col = n0 + t * 16 + ln;
      xs[m][col] = acc[t][v] + bo[col] + query[(r0 + m) * D_ + col];
    }
  }
  __syncthreads();

  // LayerNorm: wave w normalizes rows 2w and 2w+1
  #pragma unroll
  for (int rr = 0; rr < 2; ++rr) {
    int r = wave * 2 + rr;
    float s = 0.f, sq = 0.f;
    for (int c = lane; c < D_; c += 32) {
      float x = xs[r][c];
      s += x; sq += x * x;
    }
    s  = wave_sum(s);
    sq = wave_sum(sq);
    float mean = s * (1.0f / D_);
    float var  = sq * (1.0f / D_) - mean * mean;
    float rs   = rsqrtf(var + 1e-5f);
    for (int c = lane; c < D_; c += 32)
      Yout[(r0 + r) * D_ + c] = (xs[r][c] - mean) * rs * ln_w[c] + ln_b[c];
  }
}

// ---------------------------------------------------------------------------
// launch: inputs = query, key, values, lens(unused), Wq, bq, Wv, bv, Wo, bo,
//         gammas, ln_w, ln_b
// ---------------------------------------------------------------------------
extern "C" void kernel_launch(void* const* d_in, const int* in_sizes, int n_in,
                              void* d_out, int out_size, void* d_ws, size_t ws_size,
                              hipStream_t stream) {
  const float* query  = (const float*)d_in[0];
  const float* key    = (const float*)d_in[1];
  const float* values = (const float*)d_in[2];
  // d_in[3] = lens : unused by the reference computation
  const float* Wq = (const float*)d_in[4];
  const float* bq = (const float*)d_in[5];
  const float* Wv = (const float*)d_in[6];
  const float* bv = (const float*)d_in[7];
  const float* Wo = (const float*)d_in[8];
  const float* bo = (const float*)d_in[9];
  const float* gammas = (const float*)d_in[10];
  const float* ln_w   = (const float*)d_in[11];
  const float* ln_b   = (const float*)d_in[12];

  float* y_out      = (float*)d_out;                 // [B,S,D]
  float* scores_out = y_out + (size_t)B_ * S_ * D_;  // [B,H,S,S]

  // workspace layout (f16), total ~33.5 MB
  const size_t WN = (size_t)D_ * D_;
  const size_t QN = (size_t)B_ * H_ * S_ * DK_;
  _Float16* WqT16  = (_Float16*)d_ws;
  _Float16* WvT16  = WqT16 + WN;
  _Float16* WoT16  = WvT16 + WN;
  _Float16* q16    = WoT16 + WN;
  _Float16* k16    = q16 + QN;
  _Float16* vT16   = k16 + QN;
  _Float16* attn16 = vT16 + QN;

  // 0) weights -> k-major f16
  cvtT_kernel<<<(int)(WN / 256), 256, 0, stream>>>(Wq, WqT16);
  cvtT_kernel<<<(int)(WN / 256), 256, 0, stream>>>(Wv, WvT16);
  cvtT_kernel<<<(int)(WN / 256), 256, 0, stream>>>(Wo, WoT16);

  // 1) projections (k shares Wq/bq: kq_same=True); V stored transposed
  dim3 pg(D_ / 64, (B_ * S_) / 128);
  proj_kernel<false><<<pg, 256, 0, stream>>>(query,  WqT16, bq, q16);
  proj_kernel<false><<<pg, 256, 0, stream>>>(key,    WqT16, bq, k16);
  proj_kernel<true ><<<pg, 256, 0, stream>>>(values, WvT16, bv, vT16);

  // 2) attention: one block per (b,h,row-tile); 64 KB dynamic LDS
  attn_kernel<<<B_ * H_ * (S_ / 16), 128, 16 * S_ * sizeof(float), stream>>>(
      q16, k16, vT16, gammas, scores_out, attn16);

  // 3) output projection + residual + layernorm
  outproj_ln_kernel<<<(B_ * S_) / 16, 256, 0, stream>>>(
      attn16, WoT16, bo, query, ln_w, ln_b, y_out);
}
